// MDLABlock_67473936220611
// MI455X (gfx1250) — matrix operand order repaired and checked by reading, not compiled, not run on hardware
//
#include <hip/hip_runtime.h>
#include <math.h>

typedef __bf16 bf16;
typedef __attribute__((ext_vector_type(16))) __bf16 v16bf;
typedef __attribute__((ext_vector_type(8)))  __bf16 v8bf;
typedef __attribute__((ext_vector_type(8)))  float  v8f;

#define NVOX 32768          // 32*32*32 voxels
#define CDIM 192

#define EPI_BF16          0
#define EPI_BIAS_RES_F32  1
#define EPI_GELU_BF16     2
#define EPI_OUT_TRANSPOSE 3

// ---------------------------------------------------------------------------
// WMMA GEMM: Y[N][COUT] = epi( X[N][CIN](bf16) * Wt[COUT][CIN]^T )
// One wave = 16(M) x 64(N) tile: one A fragment feeds 4 B fragments / 4
// accumulators per K-step (4 back-to-back v_wmma, 2.5 b128 loads per wmma).
// Fragments: A 16x32, B 32x16 with k = 16*half + s (contiguous, same in A and B); f32 C 16x16 per ISA.
// ---------------------------------------------------------------------------
template<int CIN, int COUT, int EPI>
__global__ __launch_bounds__(256)
void gemm_wmma(const bf16* __restrict__ X, const bf16* __restrict__ Wt,
               const float* __restrict__ bias, const float* __restrict__ residual,
               bf16* __restrict__ outb, float* __restrict__ outf)
{
    const int lane = threadIdx.x & 31;
    const int wave = threadIdx.x >> 5;
    const int tile = blockIdx.x * 8 + wave;
    const int ntn  = COUT / 64;                 // 64-wide N tiles
    if (tile >= (NVOX / 16) * ntn) return;
    const int m0 = (tile / ntn) * 16;
    const int n0 = (tile % ntn) * 64;

    const int half = lane >> 4;      // 0: lanes 0-15, 1: lanes 16-31
    const int l15  = lane & 15;

    // A: lane(l15,half) holds row m0+l15, K-chunk [16*half,+16) contiguous
    const bf16* xrow = X  + (size_t)(m0 + l15) * CIN;
    // B: lane(l15,half) holds col n0+16*t+l15, K-chunk [16*half,+16) contiguous
    const bf16* wrow = Wt + (size_t)(n0 + l15) * CIN + 16 * half;

    v8f acc0 = {}, acc1 = {}, acc2 = {}, acc3 = {};
#pragma unroll
    for (int k0 = 0; k0 < CIN; k0 += 32) {
        if (k0 + 32 < CIN) {
            __builtin_prefetch(xrow + k0 + 32, 0, 3);   // global_prefetch_b8
            __builtin_prefetch(wrow + k0 + 32, 0, 3);
        }
        const v8bf alo = *(const v8bf*)(xrow + k0 + 16 * half);
        const v8bf ahi = *(const v8bf*)(xrow + k0 + 16 * half + 8);
        v16bf a;
#pragma unroll
        for (int i = 0; i < 8; ++i) { a[i] = alo[i]; a[i + 8] = ahi[i]; }
        const v16bf b0 = *(const v16bf*)(wrow + k0);
        const v16bf b1 = *(const v16bf*)(wrow + k0 + 16 * (size_t)CIN);
        const v16bf b2 = *(const v16bf*)(wrow + k0 + 32 * (size_t)CIN);
        const v16bf b3 = *(const v16bf*)(wrow + k0 + 48 * (size_t)CIN);
        acc0 = __builtin_amdgcn_wmma_f32_16x16x32_bf16(false, a, false, b0, (short)0, acc0, false, false);
        acc1 = __builtin_amdgcn_wmma_f32_16x16x32_bf16(false, a, false, b1, (short)0, acc1, false, false);
        acc2 = __builtin_amdgcn_wmma_f32_16x16x32_bf16(false, a, false, b2, (short)0, acc2, false, false);
        acc3 = __builtin_amdgcn_wmma_f32_16x16x32_bf16(false, a, false, b3, (short)0, acc3, false, false);
    }

    // C/D layout: acc[j] -> (row = m0 + j + 8*half, col = n0 + 16*t + l15)
    v8f accs[4] = {acc0, acc1, acc2, acc3};
#pragma unroll
    for (int t = 0; t < 4; ++t) {
        const int col = n0 + 16 * t + l15;
#pragma unroll
        for (int j = 0; j < 8; ++j) {
            const int row = m0 + j + 8 * half;
            float v = accs[t][j];
            if (EPI == EPI_BF16) {
                outb[(size_t)row * COUT + col] = (bf16)v;
            } else if (EPI == EPI_BIAS_RES_F32) {
                v += bias[col] + residual[(size_t)row * COUT + col];
                outf[(size_t)row * COUT + col] = v;
            } else if (EPI == EPI_GELU_BF16) {
                v += bias[col];
                v = 0.5f * v * (1.0f + erff(v * 0.70710678118654752f));  // exact GELU
                outb[(size_t)row * COUT + col] = (bf16)v;
            } else {  // EPI_OUT_TRANSPOSE: [n=(z*32+y)*32+x][c] -> out[c][y][x][z]
                v += bias[col] + residual[(size_t)row * COUT + col];
                const int xw = row & 31, yh = (row >> 5) & 31, zd = row >> 10;
                outf[(((size_t)col * 32 + yh) * 32 + xw) * 32 + zd] = v;
            }
        }
    }
}

// ---------------------------------------------------------------------------
// LN1 fused with input transpose: x[C][H][W][D] -> h[n][C] (f32) + ln[n][C] (bf16)
// One wave32 per voxel; shuffle reductions over the 192-channel row.
// ---------------------------------------------------------------------------
__global__ __launch_bounds__(256)
void ln_transpose_kernel(const float* __restrict__ x, const float* __restrict__ g,
                         const float* __restrict__ be, float* __restrict__ h,
                         bf16* __restrict__ ln)
{
    const int n    = blockIdx.x * 8 + (threadIdx.x >> 5);
    const int lane = threadIdx.x & 31;
    const int xw = n & 31, yh = (n >> 5) & 31, zd = n >> 10;
    float v[6]; float s = 0.f;
#pragma unroll
    for (int i = 0; i < 6; ++i) {
        const int c = lane + i * 32;
        v[i] = x[(((size_t)c * 32 + yh) * 32 + xw) * 32 + zd];
        s += v[i];
    }
#pragma unroll
    for (int off = 16; off > 0; off >>= 1) s += __shfl_xor(s, off, 32);
    const float mu = s * (1.0f / 192.0f);
    float vs = 0.f;
#pragma unroll
    for (int i = 0; i < 6; ++i) { const float d = v[i] - mu; vs += d * d; }
#pragma unroll
    for (int off = 16; off > 0; off >>= 1) vs += __shfl_xor(vs, off, 32);
    const float inv = rsqrtf(vs * (1.0f / 192.0f) + 1e-5f);
#pragma unroll
    for (int i = 0; i < 6; ++i) {
        const int c = lane + i * 32;
        h[(size_t)n * CDIM + c]  = v[i];
        ln[(size_t)n * CDIM + c] = (bf16)((v[i] - mu) * inv * g[c] + be[c]);
    }
}

// Plain LN: hin[n][192] (f32) -> ln[n][192] (bf16)
__global__ __launch_bounds__(256)
void ln_kernel(const float* __restrict__ hin, const float* __restrict__ g,
               const float* __restrict__ be, bf16* __restrict__ ln)
{
    const int n    = blockIdx.x * 8 + (threadIdx.x >> 5);
    const int lane = threadIdx.x & 31;
    float v[6]; float s = 0.f;
#pragma unroll
    for (int i = 0; i < 6; ++i) { v[i] = hin[(size_t)n * CDIM + lane + i * 32]; s += v[i]; }
#pragma unroll
    for (int off = 16; off > 0; off >>= 1) s += __shfl_xor(s, off, 32);
    const float mu = s * (1.0f / 192.0f);
    float vs = 0.f;
#pragma unroll
    for (int i = 0; i < 6; ++i) { const float d = v[i] - mu; vs += d * d; }
#pragma unroll
    for (int off = 16; off > 0; off >>= 1) vs += __shfl_xor(vs, off, 32);
    const float inv = rsqrtf(vs * (1.0f / 192.0f) + 1e-5f);
#pragma unroll
    for (int i = 0; i < 6; ++i) {
        const int c = lane + i * 32;
        ln[(size_t)n * CDIM + c] = (bf16)((v[i] - mu) * inv * g[c] + be[c]);
    }
}

// ---------------------------------------------------------------------------
// Dilated 3x3x3 neighborhood attention. One thread = (voxel, head).
// qkv[n][576] bf16 (q:0..191, k:192..383, v:384..575, each group-major).
// OOB neighbors behave exactly like the reference's zero padding:
// score = 0 enters the softmax, value contributes nothing.
// ---------------------------------------------------------------------------
__global__ __launch_bounds__(256)
void attn_kernel(const bf16* __restrict__ qkv, bf16* __restrict__ out)
{
    const int t    = blockIdx.x * blockDim.x + threadIdx.x;   // N*6 threads
    const int n    = t & (NVOX - 1);
    const int head = t >> 15;
    const int grp  = head >> 1;          // dilation group 0,1,2
    const int hg   = head & 1;           // head within group
    const int r    = grp + 1;            // dilation 1,2,3
    const int co   = grp * 64 + hg * 32; // channel offset within 192
    const int xw = n & 31, yh = (n >> 5) & 31, zd = n >> 10;

    const bf16* qp = qkv + (size_t)n * 576 + co;
    float q[32];
#pragma unroll
    for (int i = 0; i < 4; ++i) {
        const v8bf qq = *(const v8bf*)(qp + i * 8);
#pragma unroll
        for (int e = 0; e < 8; ++e) q[i * 8 + e] = (float)qq[e];
    }

    float s[27];
    float smax = -1e30f;
#pragma unroll
    for (int j = 0; j < 27; ++j) {
        const int a = j / 9, b = (j / 3) % 3, c = j % 3;
        const int zz = zd + (a - 1) * r, yy = yh + (b - 1) * r, xx = xw + (c - 1) * r;
        float dot = 0.f;
        if ((unsigned)zz < 32u && (unsigned)yy < 32u && (unsigned)xx < 32u) {
            const int nn = (zz * 32 + yy) * 32 + xx;
            const bf16* kp = qkv + (size_t)nn * 576 + 192 + co;
#pragma unroll
            for (int i = 0; i < 4; ++i) {
                const v8bf kk = *(const v8bf*)(kp + i * 8);
#pragma unroll
                for (int e = 0; e < 8; ++e) dot += q[i * 8 + e] * (float)kk[e];
            }
            dot *= 0.17677669529663687f;   // 1/sqrt(32)
        }
        s[j] = dot;
        smax = fmaxf(smax, dot);
    }
    float sum = 0.f;
#pragma unroll
    for (int j = 0; j < 27; ++j) { s[j] = __expf(s[j] - smax); sum += s[j]; }
    const float inv = 1.f / sum;

    float o[32] = {};
#pragma unroll
    for (int j = 0; j < 27; ++j) {
        const int a = j / 9, b = (j / 3) % 3, c = j % 3;
        const int zz = zd + (a - 1) * r, yy = yh + (b - 1) * r, xx = xw + (c - 1) * r;
        if ((unsigned)zz < 32u && (unsigned)yy < 32u && (unsigned)xx < 32u) {
            const int nn = (zz * 32 + yy) * 32 + xx;
            const float aw = s[j] * inv;
            const bf16* vp = qkv + (size_t)nn * 576 + 384 + co;
#pragma unroll
            for (int i = 0; i < 4; ++i) {
                const v8bf vv = *(const v8bf*)(vp + i * 8);
#pragma unroll
                for (int e = 0; e < 8; ++e) o[i * 8 + e] += aw * (float)vv[e];
            }
        }
    }
    bf16* op = out + (size_t)n * CDIM + co;
#pragma unroll
    for (int i = 0; i < 4; ++i) {
        v8bf ov;
#pragma unroll
        for (int e = 0; e < 8; ++e) ov[e] = (bf16)o[i * 8 + e];
        *(v8bf*)(op + i * 8) = ov;
    }
}

// ---------------------------------------------------------------------------
// Weight prep: fp32 -> bf16, transposed to [out][in] so WMMA B-fragments are
// contiguous 32B loads. qkv_w is already [576][192]=[out][in].
// ---------------------------------------------------------------------------
__global__ __launch_bounds__(256)
void prep_weights(const float* __restrict__ qkv_w, const float* __restrict__ proj_w,
                  const float* __restrict__ w1,    const float* __restrict__ w2,
                  bf16* __restrict__ qkvb, bf16* __restrict__ projt,
                  bf16* __restrict__ w1t,  bf16* __restrict__ w2t)
{
    const int i = blockIdx.x * 256 + threadIdx.x;
    if (i < 576 * 192) qkvb[i] = (bf16)qkv_w[i];
    if (i < 192 * 192) { const int o = i / 192, c = i % 192; projt[i] = (bf16)proj_w[c * 192 + o]; }
    if (i < 768 * 192) { const int o = i / 192, c = i % 192; w1t[i]   = (bf16)w1[c * 768 + o]; }
    if (i < 192 * 768) { const int o = i / 768, c = i % 768; w2t[i]   = (bf16)w2[c * 192 + o]; }
}

// ---------------------------------------------------------------------------
extern "C" void kernel_launch(void* const* d_in, const int* in_sizes, int n_in,
                              void* d_out, int out_size, void* d_ws, size_t ws_size,
                              hipStream_t stream) {
    (void)in_sizes; (void)n_in; (void)out_size; (void)ws_size;
    const float* x      = (const float*)d_in[0];
    const float* qkv_w  = (const float*)d_in[1];
    const float* proj_w = (const float*)d_in[2];
    const float* proj_b = (const float*)d_in[3];
    const float* g1     = (const float*)d_in[4];
    const float* be1    = (const float*)d_in[5];
    const float* g2     = (const float*)d_in[6];
    const float* be2    = (const float*)d_in[7];
    const float* w1     = (const float*)d_in[8];
    const float* b1     = (const float*)d_in[9];
    const float* w2     = (const float*)d_in[10];
    const float* b2     = (const float*)d_in[11];
    float* outp = (float*)d_out;

    char* ws = (char*)d_ws;
    float* ws_h    = (float*)(ws + 0);           // [N][192] f32   25.2 MB
    bf16*  ws_ln   = (bf16*) (ws + 25165824);    // [N][192] bf16  12.6 MB
    bf16*  ws_qkv  = (bf16*) (ws + 37748736);    // [N][576] bf16  37.7 MB
    bf16*  ws_attn = (bf16*) (ws + 75497472);    // [N][192] bf16  12.6 MB
    float* ws_h2   = (float*)(ws + 88080384);    // [N][192] f32   25.2 MB
    bf16*  ws_ln2  = (bf16*) (ws + 113246208);   // [N][192] bf16  12.6 MB
    bf16*  ws_mid  = (bf16*) (ws + 125829120);   // [N][768] bf16  50.3 MB
    bf16*  qkvb    = (bf16*) (ws + 176160768);   // [576][192]
    bf16*  projt   = (bf16*) (ws + 176381952);   // [192][192] (out-major)
    bf16*  w1t     = (bf16*) (ws + 176455680);   // [768][192] (out-major)
    bf16*  w2t     = (bf16*) (ws + 176750592);   // [192][768] (out-major)

    // 1) weights -> bf16 (transposed to [out][in])
    prep_weights<<<576, 256, 0, stream>>>(qkv_w, proj_w, w1, w2, qkvb, projt, w1t, w2t);
    // 2) transpose + LN1
    ln_transpose_kernel<<<NVOX / 8, 256, 0, stream>>>(x, g1, be1, ws_h, ws_ln);
    // 3) QKV GEMM [N,192]x[192,576]: 2048 * 9 tiles / 8 waves
    gemm_wmma<192, 576, EPI_BF16><<<2304, 256, 0, stream>>>(
        ws_ln, qkvb, nullptr, nullptr, ws_qkv, nullptr);
    // 4) dilated neighborhood attention (N*6 threads)
    attn_kernel<<<(NVOX * 6) / 256, 256, 0, stream>>>(ws_qkv, ws_attn);
    // 5) projection + bias + residual -> h2 (f32): 2048 * 3 tiles / 8
    gemm_wmma<192, 192, EPI_BIAS_RES_F32><<<768, 256, 0, stream>>>(
        ws_attn, projt, proj_b, ws_h, nullptr, ws_h2);
    // 6) LN2
    ln_kernel<<<NVOX / 8, 256, 0, stream>>>(ws_h2, g2, be2, ws_ln2);
    // 7) MLP up + exact GELU: 2048 * 12 tiles / 8
    gemm_wmma<192, 768, EPI_GELU_BF16><<<3072, 256, 0, stream>>>(
        ws_ln2, w1t, b1, nullptr, ws_mid, nullptr);
    // 8) MLP down + bias + residual + transpose-store: 2048 * 3 tiles / 8
    gemm_wmma<768, 192, EPI_OUT_TRANSPOSE><<<768, 256, 0, stream>>>(
        ws_mid, w2t, b2, ws_h2, nullptr, outp);
}
